// MultiHeadMomentumLayer_37022618091929
// MI455X (gfx1250) — matrix operand order repaired and checked by reading, not compiled, not run on hardware
//
#include <hip/hip_runtime.h>
#include <hip/hip_bf16.h>

// ---------------------------------------------------------------------------
// MultiHeadMomentumLayer for MI455X (gfx1250, wave32, WMMA)
//
// Shapes: B=4 S=2048 D=1024 H=8 HD=128 HH=512 E=8 K=2, N = B*S*H = 65536
// All GEMMs run on v_wmma_f32_16x16x32_bf16 (fp32 accumulate).
// Expert weights are staged per-block into LDS with
// global_load_async_to_lds_b128 (ASYNCcnt) -- one expert = 256 KB, fits the
// 320 KB WGP LDS; B-fragments are then fed to WMMA via ds_load_b128.
// ---------------------------------------------------------------------------

typedef unsigned short u16_t;
typedef __attribute__((ext_vector_type(16))) __bf16    v16bf;
typedef __attribute__((ext_vector_type(8)))  float     v8f;
typedef __attribute__((ext_vector_type(8)))  u16_t     ush8;
typedef __attribute__((ext_vector_type(16))) u16_t     ush16;

#define NTOK   65536L          // B*S*H sub-tokens
#define MROWS  8192L           // B*S token rows for split/merge GEMMs
#define DDIM   1024
#define HD     128
#define HH     512
#define NEXP   8
#define MERGED_ELEMS (MROWS * DDIM)   // 8388608

static __device__ __forceinline__ u16_t f2bf(float f) {
  unsigned u = __builtin_bit_cast(unsigned, f);
  unsigned r = u + 0x7FFFu + ((u >> 16) & 1u);   // round-to-nearest-even
  return (u16_t)(r >> 16);
}

static __device__ __forceinline__ v16bf mk16(ush8 lo, ush8 hi) {
  ush16 u;
#pragma unroll
  for (int i = 0; i < 8; ++i) { u[i] = lo[i]; u[8 + i] = hi[i]; }
  return __builtin_bit_cast(v16bf, u);
}

static __device__ __forceinline__ v8f wmma_bf16(v16bf a, v16bf b, v8f c) {
  // D = A(16x32 bf16) * B(32x16 bf16) + C(16x16 f32)
  return __builtin_amdgcn_wmma_f32_16x16x32_bf16(
      false, a, false, b, (short)0, c, false, false);
}

// ---------------------------------------------------------------------------
// f32 -> bf16 streaming convert (for x)
// ---------------------------------------------------------------------------
__global__ void k_f32_to_bf16(const float* __restrict__ src,
                              u16_t* __restrict__ dst, long n) {
  long i = (long)blockIdx.x * blockDim.x + threadIdx.x;
  long stride = (long)gridDim.x * blockDim.x;
  for (; i < n; i += stride) dst[i] = f2bf(src[i]);
}

// ---------------------------------------------------------------------------
// Pre-swizzle a row-major f32 weight [nmat][K][N] into per-lane WMMA
// B-fragment order (bf16): fragment(mat,nt,kc) is one 32(K) x 16(N) tile;
// lane l supplies B[k][n] with n = l&15, s = 2j + t, k = (l<16?0:8) + s + (s<8?0:8).
// Each lane's 16 bf16 are contiguous (two b128 loads in the GEMM).
// ---------------------------------------------------------------------------
__global__ void k_prep_bfrag(const float* __restrict__ src, u16_t* __restrict__ dst,
                             int K, int N, int nmat, long matStride) {
  long tid = (long)blockIdx.x * blockDim.x + threadIdx.x;
  int KC = K / 32, NT = N / 16;
  long total = (long)nmat * KC * NT * 32;
  if (tid >= total) return;
  int lane = (int)(tid & 31);
  long r = tid >> 5;
  int kc = (int)(r % KC); r /= KC;
  int nt = (int)(r % NT); r /= NT;
  int mat = (int)r;
  int n = nt * 16 + (lane & 15);
  int kbase = kc * 32 + ((lane < 16) ? 0 : 8);
  long off = ((((long)mat * NT + nt) * KC + kc) * 32 + lane) * 16;
#pragma unroll
  for (int j = 0; j < 8; ++j)
#pragma unroll
    for (int t = 0; t < 2; ++t) {
      int k = kbase + 2 * j + t + ((j < 4) ? 0 : 8);
      dst[off + j * 2 + t] = f2bf(src[mat * matStride + (long)k * N + n]);
    }
}

// ---------------------------------------------------------------------------
// GEMM  out[M,1024] = A_bf16[M,1024] @ Bfrag + bias ; M = 8192.
// One wave -> 16x64 output tile (4 N-tiles), K-loop of 32 chunks,
// software-pipelined one K-chunk ahead so loads overlap WMMA issue.
// ---------------------------------------------------------------------------
__global__ void __launch_bounds__(256)
k_gemm_bias(const u16_t* __restrict__ Abf, const u16_t* __restrict__ Bfrag,
            const float* __restrict__ bias, float* __restrict__ outF,
            u16_t* __restrict__ outB) {
  const int KC = 32;          // 1024 / 32
  const int COLG = 16;        // 64 N-tiles / 4 per wave
  int lane = threadIdx.x & 31;
  long wid = (long)blockIdx.x * (blockDim.x >> 5) + (threadIdx.x >> 5);
  int cg = (int)(wid % COLG);
  long rt = wid / COLG;
  if (rt >= MROWS / 16) return;

  int lm = lane & 15;
  bool lo = lane < 16;
  long m = rt * 16 + lm;
  const u16_t* arow = Abf + m * DDIM + (lo ? 0 : 8);

  auto loadA = [&](int kc) {
    return mk16(*(const ush8*)(arow + kc * 32),
                *(const ush8*)(arow + kc * 32 + 16));
  };
  auto loadB = [&](int kc, int i) {
    const u16_t* bp = Bfrag + (((long)(cg * 4 + i) * KC + kc) * 32 + lane) * 16;
    return mk16(*(const ush8*)bp, *(const ush8*)(bp + 8));
  };

  v8f c[4] = {};
  v16bf a_cur = loadA(0), a_nxt;
  v16bf b_cur[4], b_nxt[4];
#pragma unroll
  for (int i = 0; i < 4; ++i) b_cur[i] = loadB(0, i);

#pragma unroll 2
  for (int kc = 0; kc < KC; ++kc) {
    int kn = (kc + 1 < KC) ? kc + 1 : kc;
    a_nxt = loadA(kn);                      // prefetch next K-chunk
#pragma unroll
    for (int i = 0; i < 4; ++i) b_nxt[i] = loadB(kn, i);
#pragma unroll
    for (int i = 0; i < 4; ++i) c[i] = wmma_bf16(a_cur, b_cur[i], c[i]);
    a_cur = a_nxt;
#pragma unroll
    for (int i = 0; i < 4; ++i) b_cur[i] = b_nxt[i];
  }

#pragma unroll
  for (int i = 0; i < 4; ++i) {
    int n = (cg * 4 + i) * 16 + lm;
    float bv = bias[n];
#pragma unroll
    for (int v = 0; v < 8; ++v) {
      long row = rt * 16 + (lo ? v : v + 8);
      float val = c[i][v] + bv;
      outF[row * DDIM + n] = val;
      if (outB) outB[row * DDIM + n] = f2bf(val);
    }
  }
}

// ---------------------------------------------------------------------------
// Gating: logits = sub @ Wg + bg ; top-2 softmax -> dense gates [N, 8]
// ---------------------------------------------------------------------------
__global__ void k_gate(const float* __restrict__ sub, const float* __restrict__ Wg,
                       const float* __restrict__ bg, float* __restrict__ gates) {
  __shared__ float sW[HD * NEXP];
  __shared__ float sb[NEXP];
  for (int i = threadIdx.x; i < HD * NEXP; i += blockDim.x) sW[i] = Wg[i];
  if (threadIdx.x < NEXP) sb[threadIdx.x] = bg[threadIdx.x];
  __syncthreads();

  long t = (long)blockIdx.x * blockDim.x + threadIdx.x;
  if (t >= NTOK) return;
  float lg[NEXP];
#pragma unroll
  for (int e = 0; e < NEXP; ++e) lg[e] = sb[e];
  const float* row = sub + t * HD;
  for (int k = 0; k < HD; ++k) {
    float sv = row[k];
#pragma unroll
    for (int e = 0; e < NEXP; ++e) lg[e] += sv * sW[k * NEXP + e];
  }
  int i1 = 0; float v1 = lg[0];
#pragma unroll
  for (int e = 1; e < NEXP; ++e) if (lg[e] > v1) { v1 = lg[e]; i1 = e; }
  int i2 = -1; float v2 = -3.4e38f;
#pragma unroll
  for (int e = 0; e < NEXP; ++e)
    if (e != i1 && lg[e] > v2) { v2 = lg[e]; i2 = e; }
  float g1 = 1.0f / (1.0f + __expf(v2 - v1));
  float* g = gates + t * NEXP;
#pragma unroll
  for (int e = 0; e < NEXP; ++e) g[e] = 0.0f;
  g[i1] = g1;
  g[i2] = 1.0f - g1;
}

// ---------------------------------------------------------------------------
// Dense expert ensemble, LDS-resident weights.
//
// Per expert iteration the whole block DMAs that expert's swizzled Wi (128KB)
// + Wo (128KB) into LDS with global_load_async_to_lds_b128 (ASYNCcnt), then
// every wave runs its 16-token tile:
//   GEMM1 (sub 16x128 @ Wi chunk) -> bias+ReLU -> bf16 -> LDS h-buffer
//   D-fragment -> A-fragment re-layout through LDS (s_wait_dscnt fence)
//   GEMM2 accumulate, gate-scale per row.
// 2048 v_wmma per wave; all B-fragments come from ds_load_b128.
// Dynamic LDS: 128K + 128K + 10K = 266 KB (within the 320 KB WGP limit).
// ---------------------------------------------------------------------------
__global__ void __launch_bounds__(256)
k_experts(const u16_t* __restrict__ subB, const float* __restrict__ gates,
          const u16_t* __restrict__ wiF, const float* __restrict__ bi,
          const u16_t* __restrict__ woF, const float* __restrict__ bo,
          float* __restrict__ accOut) {
  extern __shared__ u16_t smem[];
  u16_t* wiLds = smem;               // 65536 u16 = 128 KB (one expert's Wi)
  u16_t* woLds = smem + 65536;       // 65536 u16 = 128 KB (one expert's Wo)
  u16_t* hb    = smem + 131072;      // 8 waves * 16 * 40 u16 h-staging

  int tid = threadIdx.x;
  int w = tid >> 5, lane = tid & 31;
  long wid = (long)blockIdx.x * 8 + w;
  long tokBase = wid * 16;
  int lm = lane & 15;
  bool loHalf = lane < 16;
  int rowOff = loHalf ? 0 : 8;
  int k0 = loHalf ? 0 : 8;
  unsigned ldsWi32 = (unsigned)(size_t)wiLds;
  unsigned ldsWo32 = (unsigned)(size_t)woLds;
  u16_t* hrow = hb + w * 640;        // this wave's 16x40 staging tile

  // A fragments of the 16-token sub tile (16x128 bf16 -> 4 K-chunks)
  v16bf aS[4];
  const u16_t* arow = subB + (tokBase + lm) * HD + k0;
#pragma unroll
  for (int kc = 0; kc < 4; ++kc)
    aS[kc] = mk16(*(const ush8*)(arow + kc * 32),
                  *(const ush8*)(arow + kc * 32 + 16));

  v8f acc[8] = {};
  for (int e = 0; e < NEXP; ++e) {
    // ---- async DMA this expert's fragment panels into LDS ----------------
    if (e) __syncthreads();          // previous expert done with LDS panels
    {
      const u16_t* gwi = wiF + (long)e * 65536;
      const u16_t* gwo = woF + (long)e * 65536;
#pragma unroll 4
      for (int c2 = 0; c2 < 32; ++c2) {         // 256 thr * 32 * 16B = 128KB
        unsigned off = (unsigned)((c2 * 256 + tid) * 16);
        asm volatile("global_load_async_to_lds_b128 %0, %1, %2"
                     :: "v"(ldsWi32 + off), "v"(off), "s"(gwi) : "memory");
        asm volatile("global_load_async_to_lds_b128 %0, %1, %2"
                     :: "v"(ldsWo32 + off), "v"(off), "s"(gwo) : "memory");
      }
      asm volatile("s_wait_asynccnt 0" ::: "memory");
      __syncthreads();
    }

    float gv[8];
#pragma unroll
    for (int v = 0; v < 8; ++v)
      gv[v] = gates[(tokBase + rowOff + v) * NEXP + e];

    v8f oc[8] = {};
    for (int hc = 0; hc < 16; ++hc) {      // 512 / 32 hidden chunks
#pragma unroll
      for (int jj = 0; jj < 2; ++jj) {     // two 16-wide GEMM1 N-tiles
        int nt = hc * 2 + jj;
        v8f hC = {};
#pragma unroll
        for (int kc = 0; kc < 4; ++kc) {
          const u16_t* bp = wiLds + (((nt * 4 + kc) * 32) + lane) * 16;
          hC = wmma_bf16(aS[kc], mk16(*(const ush8*)bp, *(const ush8*)(bp + 8)), hC);
        }
        float biv = bi[e * HH + nt * 16 + lm];
#pragma unroll
        for (int v = 0; v < 8; ++v) {
          float hv = fmaxf(hC[v] + biv, 0.0f);          // bias + ReLU
          hrow[(rowOff + v) * 40 + jj * 16 + lm] = f2bf(hv);
        }
      }
      asm volatile("s_wait_dscnt 0" ::: "memory");      // LDS RAW fence (same wave)
      // reload as A-fragment (rows in lanes, K in VGPRs)
      v16bf hA = mk16(*(const ush8*)&hrow[lm * 40 + k0],
                      *(const ush8*)&hrow[lm * 40 + k0 + 16]);
#pragma unroll
      for (int nt2 = 0; nt2 < 8; ++nt2) {
        const u16_t* bp = woLds + (((nt2 * 16 + hc) * 32) + lane) * 16;
        oc[nt2] = wmma_bf16(hA, mk16(*(const ush8*)bp, *(const ush8*)(bp + 8)), oc[nt2]);
      }
    }
#pragma unroll
    for (int nt2 = 0; nt2 < 8; ++nt2) {
      float bov = bo[e * HD + nt2 * 16 + lm];
#pragma unroll
      for (int v = 0; v < 8; ++v)
        acc[nt2][v] += gv[v] * (oc[nt2][v] + bov);
    }
  }
#pragma unroll
  for (int nt2 = 0; nt2 < 8; ++nt2)
#pragma unroll
    for (int v = 0; v < 8; ++v)
      accOut[(tokBase + rowOff + v) * HD + nt2 * 16 + lm] = acc[nt2][v];
}

// ---------------------------------------------------------------------------
// Fused momentum update + LayerNorm(HD). One wave per token (4 f32 per lane),
// wave32 shfl_xor reductions for mean/var. Writes mom_new (output #2) and
// bf16 LN activations for the merge GEMM.
// ---------------------------------------------------------------------------
__global__ void __launch_bounds__(256)
k_ln_mom(const float* __restrict__ sub, const float* __restrict__ accIn,
         const float* __restrict__ mom, const float* __restrict__ ln_g,
         const float* __restrict__ ln_b, float* __restrict__ momOut,
         u16_t* __restrict__ lnB) {
  int w = threadIdx.x >> 5, lane = threadIdx.x & 31;
  long tok = (long)blockIdx.x * 8 + w;
  long base = tok * HD + lane * 4;
  float4 s = *(const float4*)(sub + base);
  float4 a = *(const float4*)(accIn + base);
  float4 m = *(const float4*)(mom + base);
  float4 mn, o;
  mn.x = 0.7f * m.x + a.x; o.x = s.x - mn.x;
  mn.y = 0.7f * m.y + a.y; o.y = s.y - mn.y;
  mn.z = 0.7f * m.z + a.z; o.z = s.z - mn.z;
  mn.w = 0.7f * m.w + a.w; o.w = s.w - mn.w;
  *(float4*)(momOut + base) = mn;

  float sum = o.x + o.y + o.z + o.w;
  float sq = o.x * o.x + o.y * o.y + o.z * o.z + o.w * o.w;
#pragma unroll
  for (int off = 16; off > 0; off >>= 1) {
    sum += __shfl_xor(sum, off, 32);
    sq  += __shfl_xor(sq, off, 32);
  }
  float mean = sum * (1.0f / 128.0f);
  float var = sq * (1.0f / 128.0f) - mean * mean;
  float inv = rsqrtf(var + 1e-5f);
  int col = lane * 4;
  float ov[4] = {o.x, o.y, o.z, o.w};
#pragma unroll
  for (int i = 0; i < 4; ++i) {
    float y = (ov[i] - mean) * inv * ln_g[col + i] + ln_b[col + i];
    lnB[base + i] = f2bf(y);
  }
}

// ---------------------------------------------------------------------------
// Host-side orchestration (all on `stream`, graph-capture safe)
// ---------------------------------------------------------------------------
extern "C" void kernel_launch(void* const* d_in, const int* in_sizes, int n_in,
                              void* d_out, int out_size, void* d_ws, size_t ws_size,
                              hipStream_t stream) {
  (void)in_sizes; (void)n_in; (void)out_size; (void)ws_size;
  const float* x        = (const float*)d_in[0];
  const float* momentum = (const float*)d_in[1];
  const float* W_split  = (const float*)d_in[2];
  const float* b_split  = (const float*)d_in[3];
  const float* W_merge  = (const float*)d_in[4];
  const float* b_merge  = (const float*)d_in[5];
  const float* Wg       = (const float*)d_in[6];
  const float* bg       = (const float*)d_in[7];
  const float* Wi       = (const float*)d_in[8];
  const float* bi       = (const float*)d_in[9];
  const float* Wo       = (const float*)d_in[10];
  const float* bo       = (const float*)d_in[11];
  const float* ln_g     = (const float*)d_in[12];
  const float* ln_b     = (const float*)d_in[13];

  char* ws = (char*)d_ws;
  u16_t* xb    = (u16_t*)(ws);                       // 16 MB  x bf16
  float* subF  = (float*)(ws + 16777216);            // 32 MB  sub f32
  u16_t* subB  = (u16_t*)(ws + 50331648);            // 16 MB  sub bf16
  float* gates = (float*)(ws + 67108864);            //  2 MB  dense gates
  float* accW  = (float*)(ws + 69206016);            // 32 MB  expert acc
  u16_t* lnBuf = (u16_t*)(ws + 102760448);           // 16 MB  LN out bf16
  u16_t* wsF   = (u16_t*)(ws + 119537664);           //  2 MB  W_split frags
  u16_t* wmF   = (u16_t*)(ws + 121634816);           //  2 MB  W_merge frags
  u16_t* wiF   = (u16_t*)(ws + 123731968);           //  1 MB  Wi frags
  u16_t* woF   = (u16_t*)(ws + 124780544);           //  1 MB  Wo frags

  float* merged = (float*)d_out;
  float* momOut = (float*)d_out + MERGED_ELEMS;

  // 1) precision conversion + weight swizzling (one-shot, tiny)
  k_f32_to_bf16<<<4096, 256, 0, stream>>>(x, xb, MERGED_ELEMS);
  k_prep_bfrag<<<256, 256, 0, stream>>>(W_split, wsF, 1024, 1024, 1, 1024L * 1024);
  k_prep_bfrag<<<256, 256, 0, stream>>>(W_merge, wmF, 1024, 1024, 1, 1024L * 1024);
  k_prep_bfrag<<<128, 256, 0, stream>>>(Wi, wiF, 128, 512, 8, 128L * 512);
  k_prep_bfrag<<<128, 256, 0, stream>>>(Wo, woF, 512, 128, 8, 512L * 128);

  // 2) sub = x @ W_split + b_split  (also emits bf16 copy)
  k_gemm_bias<<<1024, 256, 0, stream>>>(xb, wsF, b_split, subF, subB);

  // 3) top-2 gating
  k_gate<<<256, 256, 0, stream>>>(subF, Wg, bg, gates);

  // 4) dense 8-expert ensemble (compute-bound core, LDS-resident weights)
  k_experts<<<512, 256, 272384, stream>>>(subB, gates, wiF, bi, woF, bo, accW);

  // 5) momentum + LayerNorm (streaming); writes mom_new output
  k_ln_mom<<<8192, 256, 0, stream>>>(subF, accW, momentum, ln_g, ln_b, momOut, lnBuf);

  // 6) merged = ln @ W_merge + b_merge  -> output #1
  k_gemm_bias<<<1024, 256, 0, stream>>>(lnBuf, wmF, b_merge, merged, nullptr);
}
